// AxialAttention2d_16544214024501
// MI455X (gfx1250) — matrix operand order repaired and checked by reading, not compiled, not run on hardware
//
#include <hip/hip_runtime.h>

// ---------------------------------------------------------------------------
// AxialAttention2d fused pipeline for MI455X (gfx1250, wave32, WMMA bf16)
//
// Shapes: B=4, CIN=128, H=W=s=128, COUT=128, GROUPS=8, GC=16, QC=8, SPAN=128
// N = B*H = 512 batched rows; OC2 = 2*COUT = 256 qkv channels.
//
// All WMMA operands are stored in LDS pre-transposed so fragment reads are
// contiguous along K (vectorize to ds_load_b128) with non-power-of-two row
// strides (136 for bf16, 132 for f32) to avoid LDS bank conflicts.
// ---------------------------------------------------------------------------

typedef __bf16 bf16_t;
typedef __attribute__((ext_vector_type(16))) __bf16 v16bf;
typedef __attribute__((ext_vector_type(8)))  float  v8f;

#define NROWS 512      // B*H
#define SDIM  128      // s == W
#define OC2   256      // 2*COUT
#define EPSV  1e-5f
#define STRB  136      // bf16 LDS row stride (272B -> dword stride 68, 16B aligned)
#define STRS  132      // f32  LDS row stride (528B -> dword stride 132, 16B aligned)

// ---------------- WMMA fragment helpers (contiguous k order, same for A and B) -------------
// A-matrix 16x32 bf16: lanes 0-15 row M=lane, K=0..15;
//                      lanes 16-31 row M=lane-16, K=16..31.
// Row-major A[m][k] = p[m*stride + k]  (contiguous in k -> b128 loads)
__device__ inline v16bf frag_a_row(const bf16_t* p, int stride, int lane) {
  int m  = lane & 15;
  int kh = (lane >> 4) ? 16 : 0;
  v16bf a;
#pragma unroll
  for (int h = 0; h < 16; ++h) {
    int k = kh + h;
    a[h] = p[m * stride + k];
  }
  return a;
}

// B-matrix 32x16 bf16: lanes 0-15 col N=lane, K=0..15; lanes 16-31 K=16..31.
// Column-stored B[k][n] = p[n*stride + k]  (contiguous in k -> b128 loads)
__device__ inline v16bf frag_b_col(const bf16_t* p, int stride, int lane) {
  int n  = lane & 15;
  int kh = (lane >> 4) ? 16 : 0;
  v16bf b;
#pragma unroll
  for (int h = 0; h < 16; ++h) b[h] = p[n * stride + kh + h];
  return b;
}

// B fragment B[k][n] = (bf16) p_f32[n*stride + k]  (contiguous f32 source, e.g. P)
__device__ inline v16bf frag_b_colf32(const float* p, int stride, int lane) {
  int n  = lane & 15;
  int kh = (lane >> 4) ? 16 : 0;
  v16bf b;
#pragma unroll
  for (int h = 0; h < 16; ++h) b[h] = (bf16_t)p[n * stride + kh + h];
  return b;
}

__device__ inline v8f wmma_bf16(v16bf a, v16bf b, v8f c) {
  return __builtin_amdgcn_wmma_f32_16x16x32_bf16(false, a, false, b, (short)0, c,
                                                 false, false);
}

__device__ inline float block_reduce_sum(float v, float* red, int tid) {
  red[tid] = v;
  __syncthreads();
#pragma unroll
  for (int off = 128; off > 0; off >>= 1) {
    if (tid < off) red[tid] += red[tid + off];
    __syncthreads();
  }
  float r = red[0];
  __syncthreads();
  return r;
}

// ---------------------------------------------------------------------------
// Kernel 0: zero a small float region (sim atomic accumulators)
// ---------------------------------------------------------------------------
__global__ void k_zero(float* __restrict__ p, int n) {
  int i = blockIdx.x * blockDim.x + threadIdx.x;
  if (i < n) p[i] = 0.f;
}

// ---------------------------------------------------------------------------
// Kernel 1: qkv[n][o][w] = conv_w[o][:] . x[n][:][w]   (bf16 WMMA, fp32 acc)
// One block per n; LDS stages X transposed [w][cin] and W [o][cin] as bf16.
// ---------------------------------------------------------------------------
__global__ __launch_bounds__(256) void k_qkv(const float* __restrict__ input,
                                             const float* __restrict__ convw,
                                             float* __restrict__ qkv) {
  const int n = blockIdx.x, b = n >> 7, h = n & 127;
  const int tid = threadIdx.x, lane = tid & 31, wv = tid >> 5;

  extern __shared__ char smem[];
  bf16_t* Xst = (bf16_t*)smem;       // [128][STRB] (w, cin)  B operand, K-contig
  bf16_t* Ws  = Xst + 128 * STRB;    // [256][STRB] (o, cin)  A operand, K-contig

  for (int t = tid; t < 128 * 128; t += 256) {
    int c = t >> 7, w = t & 127;     // coalesced global read over w
    Xst[w * STRB + c] = (bf16_t)input[(((size_t)b * 128 + c) * 128 + h) * 128 + w];
  }
  for (int t = tid; t < 256 * 128; t += 256) {
    int o = t >> 7, c = t & 127;
    Ws[o * STRB + c] = (bf16_t)convw[t];
  }
  __syncthreads();

  // 256x128 output = 16x8 tiles of 16x16; 8 waves -> 16 tiles each; K=128 -> 4 WMMAs
  for (int tile = wv; tile < 128; tile += 8) {
    int to = tile >> 3, tw = tile & 7;
    v8f acc = {};
#pragma unroll
    for (int kc = 0; kc < 4; ++kc) {
      v16bf a  = frag_a_row(Ws + to * 16 * STRB + kc * 32, STRB, lane);
      v16bf bb = frag_b_col(Xst + tw * 16 * STRB + kc * 32, STRB, lane);
      acc = wmma_bf16(a, bb, acc);
    }
    int w  = tw * 16 + (lane & 15);
    int ob = to * 16 + (lane >> 4) * 8;
#pragma unroll
    for (int r = 0; r < 8; ++r)
      qkv[((size_t)n * OC2 + (ob + r)) * SDIM + w] = acc[r];
  }
}

// ---------------------------------------------------------------------------
// Kernel 2: per-channel BN stats over [n][C][w] (axes n,w) -> scale/bias
// ---------------------------------------------------------------------------
__global__ __launch_bounds__(256) void k_chstats(const float* __restrict__ src, int C,
                                                 const float* __restrict__ gamma,
                                                 const float* __restrict__ beta,
                                                 float* __restrict__ scale,
                                                 float* __restrict__ bias) {
  const int c = blockIdx.x, tid = threadIdx.x;
  float s = 0.f, s2 = 0.f;
  for (int i = tid; i < NROWS * SDIM; i += 256) {
    int n = i >> 7, w = i & 127;
    float v = src[((size_t)n * C + c) * SDIM + w];
    s += v;
    s2 += v * v;
  }
  __shared__ float rs[256], rq[256];
  rs[tid] = s;
  rq[tid] = s2;
  __syncthreads();
#pragma unroll
  for (int off = 128; off > 0; off >>= 1) {
    if (tid < off) { rs[tid] += rs[tid + off]; rq[tid] += rq[tid + off]; }
    __syncthreads();
  }
  if (tid == 0) {
    const float cnt = (float)(NROWS * SDIM);
    float m   = rs[0] / cnt;
    float var = rq[0] / cnt - m * m;
    float inv = rsqrtf(var + EPSV);
    float g   = gamma[c];
    scale[c] = inv * g;
    bias[c]  = beta[c] - m * inv * g;
  }
}

// ---------------------------------------------------------------------------
// Kernel 3: per-(n,g) recompute QE, KE, QK and accumulate BN-sim stats.
// Aq[i][m] = q_embedded[m][i], Bk[j][m] = k_embedded[m][j]  (K-contiguous).
// ---------------------------------------------------------------------------
__global__ __launch_bounds__(256) void k_simstats(const float* __restrict__ qkv,
                                                  const float* __restrict__ qsc,
                                                  const float* __restrict__ qbi,
                                                  const float* __restrict__ rel_emb,
                                                  float* __restrict__ simSum,
                                                  float* __restrict__ simSq) {
  const int n = blockIdx.x >> 3, g = blockIdx.x & 7;
  const int tid = threadIdx.x, lane = tid & 31, wv = tid >> 5;

  extern __shared__ char smem[];
  float*  q   = (float*)smem;         // [8][128]
  float*  kk  = q + 8 * 128;          // [8][128]
  float*  re  = kk + 8 * 128;         // [16][256] rel_emb rows 0..15
  float*  red = re + 16 * 256;        // [256]
  bf16_t* Aq  = (bf16_t*)(red + 256); // [128][STRB]
  bf16_t* Bk  = Aq + 128 * STRB;      // [128][STRB]

  for (int t = tid; t < 16 * 128; t += 256) {
    int c = t >> 7, w = t & 127;
    int o = g * 32 + c;
    float v = qkv[((size_t)n * OC2 + o) * SDIM + w] * qsc[o] + qbi[o];
    if (c < 8) q[c * 128 + w] = v; else kk[(c - 8) * 128 + w] = v;
  }
  for (int t = tid; t < 16 * 255; t += 256) {
    int c = t / 255, p = t - c * 255;
    re[c * 256 + p] = rel_emb[c * 255 + p];
  }
  __syncthreads();

  // q_embedded(i,j), k_embedded(i,j); stored transposed: Aq[j][i], Bk[j][i]
  float sQ = 0.f, qQ = 0.f, sK = 0.f, qK = 0.f;
  for (int t = tid; t < 128 * 128; t += 256) {
    int i = t & 127, j = t >> 7, p = i - j + 127;   // consecutive lanes -> consecutive i
    float aq = 0.f, ak = 0.f;
#pragma unroll
    for (int c = 0; c < 8; ++c) {
      aq += q[c * 128 + i] * re[c * 256 + p];
      ak += kk[c * 128 + i] * re[(8 + c) * 256 + p];
    }
    Aq[j * STRB + i] = (bf16_t)aq;
    Bk[j * STRB + i] = (bf16_t)ak;
    sQ += aq; qQ += aq * aq; sK += ak; qK += ak * ak;
  }
  __syncthreads();

  // qk[i][j] = sum_m q_embedded(m,i) * k_embedded(m,j) = sum_m Aq[i][m]*Bk[j][m]
  float sS = 0.f, qS = 0.f;
  for (int tt = wv; tt < 64; tt += 8) {
    int ti = tt >> 3, tj = tt & 7;
    v8f acc = {};
#pragma unroll
    for (int kc = 0; kc < 4; ++kc) {
      v16bf a  = frag_a_row(Aq + ti * 16 * STRB + kc * 32, STRB, lane);
      v16bf bb = frag_b_col(Bk + tj * 16 * STRB + kc * 32, STRB, lane);
      acc = wmma_bf16(a, bb, acc);
    }
#pragma unroll
    for (int r = 0; r < 8; ++r) { float v = acc[r]; sS += v; qS += v * v; }
  }

  float t0 = block_reduce_sum(sS, red, tid);
  float t1 = block_reduce_sum(qS, red, tid);
  float t2 = block_reduce_sum(sQ, red, tid);
  float t3 = block_reduce_sum(qQ, red, tid);
  float t4 = block_reduce_sum(sK, red, tid);
  float t5 = block_reduce_sum(qK, red, tid);
  if (tid == 0) {
    atomicAdd(&simSum[g], t0);       atomicAdd(&simSq[g], t1);
    atomicAdd(&simSum[8 + g], t2);   atomicAdd(&simSq[8 + g], t3);
    atomicAdd(&simSum[16 + g], t4);  atomicAdd(&simSq[16 + g], t5);
  }
}

// ---------------------------------------------------------------------------
// Kernel 4: finalize BN-sim scale/bias for 24 channels
// ---------------------------------------------------------------------------
__global__ void k_simfin(const float* __restrict__ simSum, const float* __restrict__ simSq,
                         const float* __restrict__ gamma, const float* __restrict__ beta,
                         float* __restrict__ scale, float* __restrict__ bias) {
  int c = threadIdx.x;
  if (c >= 24) return;
  const float cnt = (float)NROWS * (float)SDIM * (float)SDIM;
  float m   = simSum[c] / cnt;
  float var = simSq[c] / cnt - m * m;
  float inv = rsqrtf(var + EPSV);
  float g   = gamma[c];
  scale[c] = inv * g;
  bias[c]  = beta[c] - m * inv * g;
}

// ---------------------------------------------------------------------------
// Kernel 5: fused attention per (n,g): recompute QE/KE/QK, BN-sim + 3-way sum,
// softmax (fp32 in LDS), attn = V*P^T (WMMA), attn_emb (VALU), write out_pre.
// ---------------------------------------------------------------------------
__global__ __launch_bounds__(256) void k_attn(const float* __restrict__ qkv,
                                              const float* __restrict__ qsc,
                                              const float* __restrict__ qbi,
                                              const float* __restrict__ rel_emb,
                                              const float* __restrict__ ssc,
                                              const float* __restrict__ sbi,
                                              float* __restrict__ out_pre) {
  const int n = blockIdx.x >> 3, g = blockIdx.x & 7;
  const int tid = threadIdx.x, lane = tid & 31, wv = tid >> 5;

  extern __shared__ char smem[];
  float*  q  = (float*)smem;          // [8][128]
  float*  kk = q + 8 * 128;           // [8][128]
  float*  re = kk + 8 * 128;          // [32][256] all rel_emb rows
  float*  S  = re + 32 * 256;         // [128][STRS] sim -> softmax probs (fp32)
  bf16_t* Aq = (bf16_t*)(S + 128 * STRS);  // [128][STRB]
  bf16_t* Bk = Aq + 128 * STRB;            // [128][STRB]
  bf16_t* Vs = Bk + 128 * STRB;            // [16][STRB]

  for (int t = tid; t < 32 * 128; t += 256) {
    int c = t >> 7, w = t & 127;
    int o = g * 32 + c;
    float v = qkv[((size_t)n * OC2 + o) * SDIM + w] * qsc[o] + qbi[o];
    if (c < 8)       q[c * 128 + w] = v;
    else if (c < 16) kk[(c - 8) * 128 + w] = v;
    else             Vs[(c - 16) * STRB + w] = (bf16_t)v;
  }
  for (int t = tid; t < 32 * 255; t += 256) {
    int c = t / 255, p = t - c * 255;
    re[c * 256 + p] = rel_emb[c * 255 + p];
  }
  __syncthreads();

  for (int t = tid; t < 128 * 128; t += 256) {
    int i = t & 127, j = t >> 7, p = i - j + 127;
    float aq = 0.f, ak = 0.f;
#pragma unroll
    for (int c = 0; c < 8; ++c) {
      aq += q[c * 128 + i] * re[c * 256 + p];
      ak += kk[c * 128 + i] * re[(8 + c) * 256 + p];
    }
    Aq[j * STRB + i] = (bf16_t)aq;   // q_embedded(i,j) stored transposed
    Bk[j * STRB + i] = (bf16_t)ak;   // k_embedded(i,j) stored transposed
  }
  __syncthreads();

  const float scQK = ssc[g],      biQK = sbi[g];
  const float scQE = ssc[8 + g],  biQE = sbi[8 + g];
  const float scKE = ssc[16 + g], biKE = sbi[16 + g];

  for (int tt = wv; tt < 64; tt += 8) {
    int ti = tt >> 3, tj = tt & 7;
    v8f acc = {};
#pragma unroll
    for (int kc = 0; kc < 4; ++kc) {
      v16bf a  = frag_a_row(Aq + ti * 16 * STRB + kc * 32, STRB, lane);
      v16bf bb = frag_b_col(Bk + tj * 16 * STRB + kc * 32, STRB, lane);
      acc = wmma_bf16(a, bb, acc);
    }
    int j  = tj * 16 + (lane & 15);
    int ib = ti * 16 + (lane >> 4) * 8;
#pragma unroll
    for (int r = 0; r < 8; ++r) {
      int i = ib + r;  // Aq/Bk reads contiguous in i -> b128
      float qe = (float)Aq[j * STRB + i], ke = (float)Bk[j * STRB + i];
      S[i * STRS + j] = acc[r] * scQK + biQK + qe * scQE + biQE + ke * scKE + biKE;
    }
  }
  __syncthreads();

  // softmax over j per row i (wave-uniform branch: waves 0..3 active)
  if (tid < 128) {
    float* row = S + tid * STRS;
    float mx = row[0];
    for (int j = 1; j < 128; ++j) mx = fmaxf(mx, row[j]);
    float sum = 0.f;
    for (int j = 0; j < 128; ++j) { float e = __expf(row[j] - mx); row[j] = e; sum += e; }
    float inv = 1.f / sum;
    for (int j = 0; j < 128; ++j) row[j] *= inv;
  }
  __syncthreads();

  // attn[c][i] = sum_j V[c][j] * P[i][j] : one 16x16 tile per wave (8 i-tiles)
  {
    int ti = wv;
    v8f acc = {};
#pragma unroll
    for (int kc = 0; kc < 4; ++kc) {
      v16bf a  = frag_a_row(Vs + kc * 32, STRB, lane);
      v16bf bb = frag_b_colf32(S + ti * 16 * STRS + kc * 32, STRS, lane);
      acc = wmma_bf16(a, bb, acc);
    }
    int i  = ti * 16 + (lane & 15);
    int cb = (lane >> 4) * 8;
#pragma unroll
    for (int r = 0; r < 8; ++r) {
      int c = cb + r;
      out_pre[((size_t)n * OC2 + (g * 32 + 2 * c)) * SDIM + i] = acc[r];
    }
  }

  // attn_emb[c][i] = sum_j P[i][j] * rel_emb[16+c][i-j+127]  (VALU)
  for (int t = tid; t < 16 * 128; t += 256) {
    int c = t >> 7, i = t & 127;
    const float* ve = re + (16 + c) * 256;
    const float* Pr = S + i * STRS;
    float acc = 0.f;
    for (int j = 0; j < 128; ++j) acc += Pr[j] * ve[i - j + 127];
    out_pre[((size_t)n * OC2 + (g * 32 + 2 * c + 1)) * SDIM + i] = acc;
  }
}

// ---------------------------------------------------------------------------
// Kernel 6: final BN(out) -> pair-sum over (2) -> transpose to (B,COUT,H,W)
// ---------------------------------------------------------------------------
__global__ __launch_bounds__(256) void k_final(const float* __restrict__ out_pre,
                                               const float* __restrict__ osc,
                                               const float* __restrict__ obi,
                                               float* __restrict__ out) {
  size_t i = (size_t)blockIdx.x * 256 + threadIdx.x;  // (b, co, h, w) flat
  if (i >= (size_t)4 * 128 * 128 * 128) return;
  int w = (int)(i & 127);
  size_t r = i >> 7;
  int h = (int)(r & 127); r >>= 7;
  int co = (int)(r & 127);
  int b = (int)(r >> 7);
  int n = b * 128 + h;
  float acc = 0.f;
#pragma unroll
  for (int t = 0; t < 2; ++t) {
    int o = co * 2 + t;
    acc += out_pre[((size_t)n * OC2 + o) * SDIM + w] * osc[o] + obi[o];
  }
  out[i] = acc;
}

// ---------------------------------------------------------------------------
extern "C" void kernel_launch(void* const* d_in, const int* in_sizes, int n_in,
                              void* d_out, int out_size, void* d_ws, size_t ws_size,
                              hipStream_t stream) {
  const float* input = (const float*)d_in[0];
  const float* convw = (const float*)d_in[1];
  const float* g_qkv = (const float*)d_in[2];
  const float* b_qkv = (const float*)d_in[3];
  const float* g_sim = (const float*)d_in[4];
  const float* b_sim = (const float*)d_in[5];
  const float* g_out = (const float*)d_in[6];
  const float* b_out = (const float*)d_in[7];
  const float* rel   = (const float*)d_in[8];

  float* ws = (float*)d_ws;
  float* qkv  = ws;                       // 512*256*128 = 16,777,216 f32
  float* outp = qkv + 16777216;           // 16,777,216 f32
  float* qsc  = outp + 16777216;          // 256
  float* qbi  = qsc + 256;                // 256
  float* ssum = qbi + 256;                // 24
  float* ssq  = ssum + 24;                // 24
  float* ssc  = ssq + 24;                 // 24
  float* sbi  = ssc + 24;                 // 24
  float* osc  = sbi + 24;                 // 256
  float* obi  = osc + 256;                // 256

  const size_t smemA = (size_t)(128 * STRB + 256 * STRB) * 2;               // ~102 KB
  const size_t smemC = (size_t)(8 * 128 + 8 * 128 + 16 * 256 + 256) * 4 +
                       (size_t)(2 * 128 * STRB) * 2;                        //  ~93 KB
  const size_t smemE = (size_t)(8 * 128 + 8 * 128 + 32 * 256 + 128 * STRS) * 4 +
                       (size_t)(2 * 128 * STRB + 16 * STRB) * 2;            // ~178 KB

  k_zero<<<1, 64, 0, stream>>>(ssum, 48);                                   // sim sums+sq
  k_qkv<<<512, 256, smemA, stream>>>(input, convw, qkv);
  k_chstats<<<256, 256, 0, stream>>>(qkv, OC2, g_qkv, b_qkv, qsc, qbi);
  k_simstats<<<4096, 256, smemC, stream>>>(qkv, qsc, qbi, rel, ssum, ssq);
  k_simfin<<<1, 32, 0, stream>>>(ssum, ssq, g_sim, b_sim, ssc, sbi);
  k_attn<<<4096, 256, smemE, stream>>>(qkv, qsc, qbi, rel, ssc, sbi, outp);
  k_chstats<<<256, 256, 0, stream>>>(outp, OC2, g_out, b_out, osc, obi);
  k_final<<<32768, 256, 0, stream>>>(outp, osc, obi, (float*)d_out);
}
